// MaxPoolingAggregator_13675175870550
// MI455X (gfx1250) — compile-verified, matrix operand order checked by reading, not run on hardware
//
#include <hip/hip_runtime.h>
#include <hip/hip_bf16.h>

// ---------------------------------------------------------------------------
// MI455X (gfx1250): bf16 WMMA + async global->LDS staging (ASYNCcnt path),
// wave32, fused bias+ReLU epilogues, fused mean-pool (ds_swizzle reduction).
// ---------------------------------------------------------------------------

typedef __attribute__((ext_vector_type(16))) __bf16 v16bf;
typedef __attribute__((ext_vector_type(8)))  __bf16 v8bf;
typedef __attribute__((ext_vector_type(8)))  float  v8f;

#define BATCH 4096
#define NNEI  25
#define DIM   128
#define HID   603
#define HIDP  640      // 603 padded to 40 tiles of 16
#define OUTD  128
#define ROWS  (BATCH * NNEI)   // 102400 = 6400 M-tiles

union FragBF { v16bf v; v8bf h[2]; };

__device__ __forceinline__ unsigned lds_offset(const void* p) {
  // Flat shared-memory addresses carry the workgroup-relative LDS offset in
  // the low 32 bits (aperture in the high bits) -> exactly what the async
  // instruction's VDST operand wants.
  return (unsigned)(size_t)p;
}

__device__ __forceinline__ void async_copy16(const void* gsrc, unsigned lofs) {
  asm volatile("global_load_async_to_lds_b128 %0, %1, off"
               :: "v"(lofs), "v"(gsrc) : "memory");
}

__device__ __forceinline__ void wait_async0() {
#if __has_builtin(__builtin_amdgcn_s_wait_asynccnt)
  __builtin_amdgcn_s_wait_asynccnt(0);
#else
  asm volatile("s_wait_asynccnt 0x0" ::: "memory");
#endif
}

__device__ __forceinline__ v8bf cvt8(const float* p) {
  v8bf r;
  #pragma unroll
  for (int i = 0; i < 8; ++i) r[i] = (__bf16)p[i];
  return r;
}

// A/B fragment for 16x16x32 bf16 WMMA. Lane l holds row/col lo=l%16, hi=l/16:
//   elems 0..7  <- K = k0 + hi*8 + {0..7};  elems 8..15 <- +16
__device__ __forceinline__ v16bf frag_bf16(const __bf16* rowp, int k0, int hi) {
  FragBF f;
  f.h[0] = *(const v8bf*)(rowp + k0 + hi * 8);
  f.h[1] = *(const v8bf*)(rowp + k0 + hi * 8 + 16);
  return f.v;
}

__device__ __forceinline__ v16bf frag_f32(const float* rowp, int k0, int hi) {
  FragBF f;
  f.h[0] = cvt8(rowp + k0 + hi * 8);
  f.h[1] = cvt8(rowp + k0 + hi * 8 + 16);
  return f.v;
}

__device__ __forceinline__ v8f wmma_bf16(v16bf a, v16bf b, v8f c) {
  return __builtin_amdgcn_wmma_f32_16x16x32_bf16(
      false, a, false, b, (short)0, c, false, false);
}

// ---------------------------------------------------------------------------
// Kernel 0: pack weights -> transposed, zero-padded bf16.
// ---------------------------------------------------------------------------
__global__ void pack_weights(const float* __restrict__ W1, const float* __restrict__ W2,
                             const float* __restrict__ NW, const float* __restrict__ SW,
                             __bf16* __restrict__ W1T, __bf16* __restrict__ W2T,
                             __bf16* __restrict__ NWT, __bf16* __restrict__ SWT) {
  int idx = blockIdx.x * 256 + threadIdx.x;
  if (idx < HIDP * DIM) {                       // W1T [HIDP][DIM]
    int n = idx / DIM, k = idx % DIM;
    W1T[idx] = (n < HID) ? (__bf16)W1[k * HID + n] : (__bf16)0.0f;
    return;
  }
  idx -= HIDP * DIM;
  if (idx < HIDP * HIDP) {                      // W2T [HIDP][HIDP]
    int n = idx / HIDP, k = idx % HIDP;
    W2T[idx] = (n < HID && k < HID) ? (__bf16)W2[k * HID + n] : (__bf16)0.0f;
    return;
  }
  idx -= HIDP * HIDP;
  if (idx < OUTD * HIDP) {                      // NWT [OUTD][HIDP]
    int n = idx / HIDP, k = idx % HIDP;
    NWT[idx] = (k < HID) ? (__bf16)NW[k * OUTD + n] : (__bf16)0.0f;
    return;
  }
  idx -= OUTD * HIDP;
  if (idx < OUTD * DIM) {                       // SWT [OUTD][DIM]
    int n = idx / DIM, k = idx % DIM;
    SWT[idx] = (__bf16)SW[k * OUTD + n];
  }
}

// ---------------------------------------------------------------------------
// Kernel 1: H1 = relu(X @ W1 + b1), X:[ROWS,DIM] f32 -> H1:[ROWS,HIDP] bf16.
// Grid 6400, block 256 (8 waves). Block = 1 M-tile, covers all 40 N-tiles
// (5 per wave). X panel (16x128 f32, rows padded to 132 floats for
// conflict-free LDS banking) staged ONCE per block via async loads.
// ---------------------------------------------------------------------------
#define XROW 132
__global__ void gemm1_relu(const float* __restrict__ X, const __bf16* __restrict__ W1T,
                           const float* __restrict__ b1, __bf16* __restrict__ H1) {
  __shared__ __attribute__((aligned(16))) float ldsX[16 * XROW];
  const int tid = threadIdx.x;
  const int lane = tid & 31, wave = tid >> 5;
  const int hi = lane >> 4, lo = lane & 15;
  const int mtile = blockIdx.x;

  // Stage 16 rows x 128 f32 = 512 chunks of 16B, 2 per thread (exact).
  #pragma unroll
  for (int c = tid; c < 16 * 32; c += 256) {
    int row = c >> 5, chunk = c & 31;
    const float* gsrc = X + (size_t)(mtile * 16 + row) * DIM + chunk * 4;
    async_copy16(gsrc, lds_offset(&ldsX[row * XROW + chunk * 4]));
  }
  wait_async0();
  __syncthreads();

  const float* arow = &ldsX[lo * XROW];
  v8f acc[5] = {};
  #pragma unroll
  for (int k = 0; k < DIM; k += 32) {
    v16bf a = frag_f32(arow, k, hi);
    #pragma unroll
    for (int t = 0; t < 5; ++t) {
      const __bf16* brow = W1T + (size_t)((wave * 5 + t) * 16 + lo) * DIM;
      acc[t] = wmma_bf16(a, frag_bf16(brow, k, hi), acc[t]);
    }
  }

  #pragma unroll
  for (int t = 0; t < 5; ++t) {
    int col = (wave * 5 + t) * 16 + lo;
    float bias = (col < HID) ? b1[col] : 0.0f;
    #pragma unroll
    for (int j = 0; j < 8; ++j) {              // C/D elem j: row m = j + 8*hi
      int row = mtile * 16 + j + 8 * hi;
      H1[(size_t)row * HIDP + col] = (__bf16)fmaxf(acc[t][j] + bias, 0.0f);
    }
  }
}

// ---------------------------------------------------------------------------
// Kernel 2: pooled[b] = mean_n relu(H1[b,n] @ W2 + b2), fused pooling.
// Grid 4096, block 256 (8 waves). A panel = 32 rows x 640 bf16 (rows padded
// to 648 -> 1296B = 324 dwords, conflict-free) staged once per block via
// async loads; clamped rows >=25 are masked in the pooling epilogue.
// ---------------------------------------------------------------------------
#define AROW 648
__global__ void gemm2_pool(const __bf16* __restrict__ H1, const __bf16* __restrict__ W2T,
                           const float* __restrict__ b2, __bf16* __restrict__ pooled) {
  __shared__ __attribute__((aligned(16))) __bf16 ldsA[32 * AROW];
  const int tid = threadIdx.x;
  const int lane = tid & 31, wave = tid >> 5;
  const int hi = lane >> 4, lo = lane & 15;
  const int b = blockIdx.x;

  // Stage 32 rows x 640 bf16 = 2560 chunks of 16B, 10 per thread (exact).
  #pragma unroll
  for (int c = tid; c < 32 * 80; c += 256) {
    int row = c / 80, chunk = c % 80;
    int grow = b * NNEI + (row < NNEI ? row : 0);      // clamp, masked later
    const __bf16* gsrc = H1 + (size_t)grow * HIDP + chunk * 8;
    async_copy16(gsrc, lds_offset(&ldsA[row * AROW + chunk * 8]));
  }
  wait_async0();
  __syncthreads();

  const __bf16* a0row = &ldsA[lo * AROW];
  const __bf16* a1row = &ldsA[(16 + lo) * AROW];
  v8f acc[10] = {};

  for (int k = 0; k < HIDP; k += 32) {
    v16bf A0 = frag_bf16(a0row, k, hi);
    v16bf A1 = frag_bf16(a1row, k, hi);
    #pragma unroll
    for (int t = 0; t < 5; ++t) {
      const __bf16* brow = W2T + (size_t)((wave * 5 + t) * 16 + lo) * HIDP;
      v16bf Bf = frag_bf16(brow, k, hi);
      acc[2 * t]     = wmma_bf16(A0, Bf, acc[2 * t]);
      acc[2 * t + 1] = wmma_bf16(A1, Bf, acc[2 * t + 1]);
    }
  }

  // Per-lane masked relu-sum over M, then cross-half (SWAPX16) add.
  #pragma unroll
  for (int t = 0; t < 5; ++t) {
    int n = (wave * 5 + t) * 16 + lo;
    float bias = (n < HID) ? b2[n] : 0.0f;
    float s = 0.0f;
    #pragma unroll
    for (int j = 0; j < 8; ++j)                 // M-tile 0: m = j+8*hi < 16
      s += fmaxf(acc[2 * t][j] + bias, 0.0f);
    #pragma unroll
    for (int j = 0; j < 8; ++j) {               // M-tile 1: m = 16+j+8*hi
      float r = fmaxf(acc[2 * t + 1][j] + bias, 0.0f);
      s += ((16 + j + 8 * hi) < NNEI) ? r : 0.0f;
    }
    int so = __builtin_amdgcn_ds_swizzle(__float_as_int(s), 0x401F); // SWAPX16
    s += __int_as_float(so);
    if (lane < 16)
      pooled[(size_t)b * HIDP + n] = (__bf16)(s * (1.0f / NNEI));
  }
}

// ---------------------------------------------------------------------------
// Kernel 3: outB = pooled @ NW + self @ SW   ([BATCH,OUTD] f32)
// ---------------------------------------------------------------------------
__global__ void gemm3(const __bf16* __restrict__ pooled, const __bf16* __restrict__ NWT,
                      const float* __restrict__ selfX, const __bf16* __restrict__ SWT,
                      float* __restrict__ outB) {
  const int lane = threadIdx.x & 31;
  const int wave = threadIdx.x >> 5;           // ntile
  const int hi = lane >> 4, lo = lane & 15;
  const int mtile = blockIdx.x;

  const __bf16* arow  = pooled + (size_t)(mtile * 16 + lo) * HIDP;
  const __bf16* browN = NWT + (size_t)(wave * 16 + lo) * HIDP;
  v8f acc = {};
  for (int k = 0; k < HIDP; k += 32)
    acc = wmma_bf16(frag_bf16(arow, k, hi), frag_bf16(browN, k, hi), acc);

  const float*  srow  = selfX + (size_t)(mtile * 16 + lo) * DIM;
  const __bf16* browS = SWT + (size_t)(wave * 16 + lo) * DIM;
  #pragma unroll
  for (int k = 0; k < DIM; k += 32)
    acc = wmma_bf16(frag_f32(srow, k, hi), frag_bf16(browS, k, hi), acc);

  #pragma unroll
  for (int j = 0; j < 8; ++j) {
    int row = mtile * 16 + j + 8 * hi;
    outB[(size_t)row * OUTD + wave * 16 + lo] = acc[j];
  }
}

// ---------------------------------------------------------------------------
// Kernel 4: out[0:S) = relu(sub+sup); out[S:2S) = relu(sub)
// ---------------------------------------------------------------------------
__global__ void combine(const float* __restrict__ subB, const float* __restrict__ supB,
                        float* __restrict__ out) {
  int i = blockIdx.x * 256 + threadIdx.x;
  float a = subB[i];
  out[i] = fmaxf(a + supB[i], 0.0f);
  out[BATCH * OUTD + i] = fmaxf(a, 0.0f);
}

// ---------------------------------------------------------------------------
// Launcher
// ---------------------------------------------------------------------------
extern "C" void kernel_launch(void* const* d_in, const int* in_sizes, int n_in,
                              void* d_out, int out_size, void* d_ws, size_t ws_size,
                              hipStream_t stream) {
  const float* self_vecs      = (const float*)d_in[0];
  const float* neigh_vecs     = (const float*)d_in[1];
  const float* sub_self_vecs  = (const float*)d_in[2];
  const float* sub_neigh_vecs = (const float*)d_in[3];
  const float* W1 = (const float*)d_in[4];
  const float* b1 = (const float*)d_in[5];
  const float* W2 = (const float*)d_in[6];
  const float* b2 = (const float*)d_in[7];
  const float* NW = (const float*)d_in[8];
  const float* SW = (const float*)d_in[9];

  char* ws = (char*)d_ws;
  size_t off = 0;
  __bf16* W1T = (__bf16*)(ws + off); off += (size_t)HIDP * DIM  * 2;
  __bf16* W2T = (__bf16*)(ws + off); off += (size_t)HIDP * HIDP * 2;
  __bf16* NWT = (__bf16*)(ws + off); off += (size_t)OUTD * HIDP * 2;
  __bf16* SWT = (__bf16*)(ws + off); off += (size_t)OUTD * DIM  * 2;
  __bf16* H1  = (__bf16*)(ws + off); off += (size_t)ROWS * HIDP * 2;   // 131 MB
  __bf16* POOL= (__bf16*)(ws + off); off += (size_t)BATCH * HIDP * 2;  // 5.2 MB
  float*  OSUB= (float*) (ws + off); off += (size_t)BATCH * OUTD * 4;
  float*  OSUP= (float*) (ws + off); off += (size_t)BATCH * OUTD * 4;

  const int packN = HIDP * DIM + HIDP * HIDP + OUTD * HIDP + OUTD * DIM;
  pack_weights<<<(packN + 255) / 256, 256, 0, stream>>>(W1, W2, NW, SW, W1T, W2T, NWT, SWT);

  // --- sub branch ---
  gemm1_relu<<<ROWS / 16, 256, 0, stream>>>(sub_neigh_vecs, W1T, b1, H1);
  gemm2_pool<<<BATCH, 256, 0, stream>>>(H1, W2T, b2, POOL);
  gemm3<<<BATCH / 16, 256, 0, stream>>>(POOL, NWT, sub_self_vecs, SWT, OSUB);

  // --- sup branch (reuses H1/POOL) ---
  gemm1_relu<<<ROWS / 16, 256, 0, stream>>>(neigh_vecs, W1T, b1, H1);
  gemm2_pool<<<BATCH, 256, 0, stream>>>(H1, W2T, b2, POOL);
  gemm3<<<BATCH / 16, 256, 0, stream>>>(POOL, NWT, self_vecs, SWT, OSUP);

  combine<<<(BATCH * OUTD) / 256, 256, 0, stream>>>(OSUB, OSUP, (float*)d_out);
}
